// StreamingTemporalConsistencyLoss_61220463837477
// MI455X (gfx1250) — compile-verified, matrix operand order checked by reading, not run on hardware
//
#include <hip/hip_runtime.h>
#include <stdint.h>

#ifndef __has_builtin
#define __has_builtin(x) 0
#endif

#if __has_builtin(__builtin_amdgcn_global_load_async_to_lds_b128)
#define HAVE_ASYNC_LDS 1
#else
#define HAVE_ASYNC_LDS 0
#endif

typedef __attribute__((ext_vector_type(16))) _Float16 v16h;
typedef __attribute__((ext_vector_type(8)))  float    v8f;
typedef __attribute__((ext_vector_type(4)))  int      i4v;
typedef __attribute__((address_space(1))) i4v* gi4p;   // global int4*
typedef __attribute__((address_space(3))) i4v* li4p;   // LDS int4*

#define BB 16
#define HW 245760            // 384*640
#define N2 491520            // 2*HW
#define NB 512               // histogram bins
#define RCAP 8.0f            // residual histogram cap (residuals are range-normalized, O(1))
#define DELTA_ 0.03f
#define KEEPF 0.6f           // 1 - TRIM
#define DIFFR 0.01f
#define LAMB 0.2f
#define NBLK 60
#define NTHR 256
#define TILE 1024

// workspace layout in 32-bit words
#define OFF_SUMS   0                      // BB*8 floats: a00,a01,a11,b0,b1
#define OFF_MINMAX (OFF_SUMS + BB*8)      // BB gmin keys, BB gmax keys (uint)
#define OFF_SPAR   (OFF_MINMAX + BB*2)    // BB*8: s,t,glo,gbinw,rr,scale,th
#define OFF_HGT    (OFF_SPAR + BB*8)      // BB*NB uint
#define OFF_SCNT   (OFF_HGT + BB*NB)      // BB uint
#define OFF_HRD    (OFF_SCNT + BB)        // BB*NB uint
#define OFF_HRA    (OFF_HRD + BB*NB)      // BB*NB uint
#define OFF_BLOSS  (OFF_HRA + BB*NB)      // BB*2 floats
#define WS_WORDS   (OFF_BLOSS + BB*2)

__device__ __forceinline__ unsigned fkey(float f) {
  unsigned u = __float_as_uint(f);
  return (u & 0x80000000u) ? ~u : (u | 0x80000000u);
}
__device__ __forceinline__ float fdec(unsigned k) {
  unsigned u = (k & 0x80000000u) ? (k & 0x7FFFFFFFu) : ~k;
  return __uint_as_float(u);
}

// ---------------- K0: init workspace ----------------
__global__ void k0_init(unsigned* wsu) {
  int i = blockIdx.x * blockDim.x + threadIdx.x;
  if (i >= WS_WORDS) return;
  unsigned v = 0u;
  if (i >= OFF_MINMAX && i < OFF_MINMAX + BB) v = 0xFFFFFFFFu;  // gmin keys
  wsu[i] = v;
}

// ---------------- K1: per-batch masked sums + gt min/max ----------------
// Block reduction of the 5 accumulators is performed by one v_wmma_f32_16x16x32_f16:
// A(16x32 f16) rows = quantities, cols = 32 pre-reduced column sums; B = ones.
__global__ void k1_sums(const float* __restrict__ pred, const float* __restrict__ gt,
                        const float* __restrict__ mask, float* __restrict__ ws,
                        unsigned* __restrict__ wsu) {
  const int b = blockIdx.y;
  const int tid = threadIdx.x;
  const float4* p4 = reinterpret_cast<const float4*>(pred + (size_t)b * N2);
  const float4* g4 = reinterpret_cast<const float4*>(gt   + (size_t)b * N2);
  const float4* m4 = reinterpret_cast<const float4*>(mask + (size_t)b * N2);
  const int nvec = N2 / 4;
  const int stride = gridDim.x * blockDim.x;

  float a00 = 0.f, a01 = 0.f, a11 = 0.f, bb0 = 0.f, bb1 = 0.f;
  float gmn = 3.0e38f, gmx = -3.0e38f;

#define ACC1(px, gx, mx) { float mm = (mx) > 0.5f ? 1.0f : 0.0f; \
    float pp = (px), gg = (gx); \
    a00 += mm * pp * pp; a01 += mm * pp; a11 += mm; \
    bb0 += mm * pp * gg; bb1 += mm * gg; \
    if (mm > 0.f) { gmn = fminf(gmn, gg); gmx = fmaxf(gmx, gg); } }

  for (int i = blockIdx.x * blockDim.x + tid; i < nvec; i += stride) {
    __builtin_prefetch(&p4[i + stride], 0, 0);
    __builtin_prefetch(&g4[i + stride], 0, 0);
    float4 p = p4[i], g = g4[i], m = m4[i];
    ACC1(p.x, g.x, m.x); ACC1(p.y, g.y, m.y);
    ACC1(p.z, g.z, m.z); ACC1(p.w, g.w, m.w);
  }
#undef ACC1

  __shared__ float stage[NTHR * 8];
  __shared__ float red[32 * 8];
  __shared__ unsigned shmn, shmx;

  stage[tid * 8 + 0] = a00; stage[tid * 8 + 1] = a01; stage[tid * 8 + 2] = a11;
  stage[tid * 8 + 3] = bb0; stage[tid * 8 + 4] = bb1;
  if (tid == 0) { shmn = 0xFFFFFFFFu; shmx = 0u; }
  __syncthreads();

  atomicMin(&shmn, fkey(gmn));
  atomicMax(&shmx, fkey(gmx));

  if (tid < 32) {
#pragma unroll
    for (int q = 0; q < 5; ++q) {
      float s = 0.f;
#pragma unroll
      for (int j = 0; j < NTHR / 32; ++j) s += stage[(tid + 32 * j) * 8 + q];
      red[tid * 8 + q] = s;
    }
  }
  __syncthreads();

  if (tid < 32) {
    const int lane = tid;
    const int m = lane & 15;          // A-matrix row this lane holds
    const int koff = (lane >= 16) ? 8 : 0;
    v16h a;
#pragma unroll
    for (int h = 0; h < 16; ++h) {
      int j = h >> 1, p = h & 1;
      int k = ((j < 4) ? (2 * j + p) : (16 + 2 * (j - 4) + p)) + koff;
      float v = (m < 5) ? red[k * 8 + m] : 0.f;
      a[h] = (_Float16)v;
    }
    v16h ones;
#pragma unroll
    for (int h = 0; h < 16; ++h) ones[h] = (_Float16)1.0f;
    v8f c = {};
    // D[m][n] = sum_k A[m][k] : 32-way reduction of 16 rows in one op
    c = __builtin_amdgcn_wmma_f32_16x16x32_f16(false, a, false, ones,
                                               (short)0, c, false, false);
    if (lane == 0) {
#pragma unroll
      for (int q = 0; q < 5; ++q)
        atomicAdd(&ws[OFF_SUMS + b * 8 + q], c[q]);
    }
  }
  if (tid == 0) {
    atomicMin(&wsu[OFF_MINMAX + b], shmn);
    atomicMax(&wsu[OFF_MINMAX + BB + b], shmx);
  }
}

// ---------------- K2: solve scale/shift, histogram params ----------------
__global__ void k2_solve(const float* __restrict__ ws, const unsigned* __restrict__ wsu,
                         float* __restrict__ spar) {
  int b = threadIdx.x;
  if (b >= BB) return;
  float a00 = ws[OFF_SUMS + b * 8 + 0], a01 = ws[OFF_SUMS + b * 8 + 1];
  float a11 = ws[OFF_SUMS + b * 8 + 2];
  float b0 = ws[OFF_SUMS + b * 8 + 3], b1 = ws[OFF_SUMS + b * 8 + 4];
  float det = a00 * a11 - a01 * a01;
  float s = 0.f, t = 0.f;
  if (det != 0.f) {
    s = (a11 * b0 - a01 * b1) / (det + 1e-6f);
    t = (-a01 * b0 + a00 * b1) / (det + 1e-6f);
  }
  unsigned kmn = wsu[OFF_MINMAX + b], kmx = wsu[OFF_MINMAX + BB + b];
  float glo = 0.f, ghi = 1.f;
  if (a11 > 0.f && kmn != 0xFFFFFFFFu) { glo = fdec(kmn); ghi = fdec(kmx); }
  float binw = (ghi - glo) / (float)NB;
  if (!(binw > 0.f)) binw = 1.0f;
  spar[b * 8 + 0] = s; spar[b * 8 + 1] = t;
  spar[b * 8 + 2] = glo; spar[b * 8 + 3] = binw;
}

// ---------------- K3: masked gt histogram (async-LDS staged) ----------------
__global__ void k3_hist_gt(const float* __restrict__ gt, const float* __restrict__ mask,
                           unsigned* __restrict__ wsu, const float* __restrict__ spar) {
  const int b = blockIdx.y;
  __shared__ unsigned hist[NB];
  __shared__ __align__(16) float sg[TILE];
  __shared__ __align__(16) float sm[TILE];
  for (int i = threadIdx.x; i < NB; i += blockDim.x) hist[i] = 0u;
  const float glo = spar[b * 8 + 2];
  const float inv = 1.0f / spar[b * 8 + 3];
  const float* gbase = gt + (size_t)b * N2;
  const float* mbase = mask + (size_t)b * N2;
  const int ntiles = N2 / TILE;
  __syncthreads();

  for (int tI = blockIdx.x; tI < ntiles; tI += gridDim.x) {
    const float* gsrc = gbase + tI * TILE + threadIdx.x * 4;
    const float* msrc = mbase + tI * TILE + threadIdx.x * 4;
#if HAVE_ASYNC_LDS
    __builtin_amdgcn_global_load_async_to_lds_b128(
        (gi4p)(__attribute__((address_space(1))) const void*)gsrc,
        (li4p)(__attribute__((address_space(3))) void*)&sg[threadIdx.x * 4], 0, 0);
    __builtin_amdgcn_global_load_async_to_lds_b128(
        (gi4p)(__attribute__((address_space(1))) const void*)msrc,
        (li4p)(__attribute__((address_space(3))) void*)&sm[threadIdx.x * 4], 0, 0);
#if __has_builtin(__builtin_amdgcn_s_wait_asynccnt)
    __builtin_amdgcn_s_wait_asynccnt(0);
#else
    asm volatile("s_wait_asynccnt 0x0" ::: "memory");
#endif
#else
    *(float4*)&sg[threadIdx.x * 4] = *(const float4*)gsrc;
    *(float4*)&sm[threadIdx.x * 4] = *(const float4*)msrc;
#endif
    __syncthreads();
    for (int i = threadIdx.x; i < TILE; i += blockDim.x) {
      if (sm[i] > 0.5f) {
        int bin = (int)((sg[i] - glo) * inv);
        bin = bin < 0 ? 0 : (bin > NB - 1 ? NB - 1 : bin);
        atomicAdd(&hist[bin], 1u);
      }
    }
    __syncthreads();
  }
  for (int i = threadIdx.x; i < NB; i += blockDim.x)
    if (hist[i]) atomicAdd(&wsu[OFF_HGT + b * NB + i], hist[i]);
}

// ---------------- K4: 5%/95% quantiles -> rr, scale, th ----------------
__global__ void k4_range(const float* __restrict__ ws, const unsigned* __restrict__ wsu,
                         float* __restrict__ spar) {
  int b = blockIdx.x;
  if (threadIdx.x != 0) return;
  float nF = ws[OFF_SUMS + b * 8 + 2];
  const unsigned* h = wsu + OFF_HGT + b * NB;
  float glo = spar[b * 8 + 2], binw = spar[b * 8 + 3];
  float posL = 0.05f * fmaxf(nF - 1.0f, 0.0f);
  float posH = 0.95f * fmaxf(nF - 1.0f, 0.0f);
  float vL = glo, vH = glo, cum = 0.f;
  bool gotL = false, gotH = false;
  for (int i = 0; i < NB; ++i) {
    float c = (float)h[i];
    if (!gotL && cum + c > posL) {
      float f = c > 0.f ? (posL - cum) / c : 0.f;
      vL = glo + ((float)i + f) * binw; gotL = true;
    }
    if (!gotH && cum + c > posH) {
      float f = c > 0.f ? (posH - cum) / c : 0.f;
      vH = glo + ((float)i + f) * binw; gotH = true;
    }
    cum += c;
    if (gotL && gotH) break;
  }
  bool valid = (nF > 0.f) && (vH - vL > 0.f);
  float rr = valid ? fmaxf(vH - vL, 1e-6f) : 1.0f;
  spar[b * 8 + 4] = rr;
  spar[b * 8 + 5] = fmaxf(rr, 1e-6f);
  spar[b * 8 + 6] = DIFFR * rr;
}

// ---------------- K5: residuals, curr_grad, residual histograms ----------------
__global__ void k5_resid(const float* __restrict__ pred, const float* __restrict__ gt,
                         const float* __restrict__ mask, const float* __restrict__ prev,
                         const float* __restrict__ spar, unsigned* __restrict__ wsu,
                         float* __restrict__ grad_out) {
  const int b = blockIdx.y;
  const int tid = threadIdx.x;
  __shared__ unsigned hrd[NB], hra[NB];
  __shared__ unsigned scnt;
  for (int i = tid; i < NB; i += blockDim.x) { hrd[i] = 0u; hra[i] = 0u; }
  if (tid == 0) scnt = 0u;
  const float s = spar[b * 8 + 0];
  const float sinv = 1.0f / spar[b * 8 + 5];
  const float th = spar[b * 8 + 6];
  const float rbin = (float)NB / RCAP;
  const float4* p0v = reinterpret_cast<const float4*>(pred + (size_t)b * N2);
  const float4* p1v = reinterpret_cast<const float4*>(pred + (size_t)b * N2 + HW);
  const float4* g0v = reinterpret_cast<const float4*>(gt + (size_t)b * N2);
  const float4* g1v = reinterpret_cast<const float4*>(gt + (size_t)b * N2 + HW);
  const float4* m0v = reinterpret_cast<const float4*>(mask + (size_t)b * N2);
  const float4* m1v = reinterpret_cast<const float4*>(mask + (size_t)b * N2 + HW);
  const float4* pvv = reinterpret_cast<const float4*>(prev + (size_t)b * HW);
  float4* gout = reinterpret_cast<float4*>(grad_out + (size_t)b * HW);
  const int nvec = HW / 4;
  const int stride = gridDim.x * blockDim.x;
  unsigned lc = 0;
  __syncthreads();

#define RES1(p0x, p1x, g0x, g1x, m0x, m1x, pvx, gox) { \
    float dp = s * ((p1x) - (p0x)); (gox) = dp; \
    if ((m0x) > 0.5f && (m1x) > 0.5f) { \
      float dg = (g1x) - (g0x); \
      if (fabsf(dg) < th) { \
        float rd = fabsf(dp - dg) * sinv; \
        float ra = fabsf(dp - (pvx)) * sinv; \
        int bd = (int)(rd * rbin); bd = bd > NB - 1 ? NB - 1 : bd; \
        int ba = (int)(ra * rbin); ba = ba > NB - 1 ? NB - 1 : ba; \
        atomicAdd(&hrd[bd], 1u); atomicAdd(&hra[ba], 1u); lc++; } } }

  for (int i = blockIdx.x * blockDim.x + tid; i < nvec; i += stride) {
    __builtin_prefetch(&p0v[i + stride], 0, 0);
    __builtin_prefetch(&g0v[i + stride], 0, 0);
    float4 p0 = p0v[i], p1 = p1v[i], g0 = g0v[i], g1 = g1v[i];
    float4 m0 = m0v[i], m1 = m1v[i], pv = pvv[i];
    float4 go;
    RES1(p0.x, p1.x, g0.x, g1.x, m0.x, m1.x, pv.x, go.x);
    RES1(p0.y, p1.y, g0.y, g1.y, m0.y, m1.y, pv.y, go.y);
    RES1(p0.z, p1.z, g0.z, g1.z, m0.z, m1.z, pv.z, go.z);
    RES1(p0.w, p1.w, g0.w, g1.w, m0.w, m1.w, pv.w, go.w);
    gout[i] = go;
  }
#undef RES1
  if (lc) atomicAdd(&scnt, lc);
  __syncthreads();
  for (int i = tid; i < NB; i += blockDim.x) {
    if (hrd[i]) atomicAdd(&wsu[OFF_HRD + b * NB + i], hrd[i]);
    if (hra[i]) atomicAdd(&wsu[OFF_HRA + b * NB + i], hra[i]);
  }
  if (tid == 0 && scnt) atomicAdd(&wsu[OFF_SCNT + b], scnt);
}

// ---------------- K6: trimmed-Huber loss from histograms ----------------
__global__ void k6_huber(const unsigned* __restrict__ wsu, float* __restrict__ ws) {
  int which = blockIdx.x, b = blockIdx.y;
  if (threadIdx.x != 0) return;
  const unsigned* h = wsu + (which ? OFF_HRA : OFF_HRD) + b * NB;
  unsigned n = wsu[OFF_SCNT + b];
  unsigned k = (unsigned)(KEEPF * (float)n);
  float loss = 0.f;
  if (k > 0) {
    const float binw = RCAP / (float)NB;
    float cum = 0.f, ssum = 0.f;
    for (int i = 0; i < NB; ++i) {
      float c = (float)h[i];
      float r = ((float)i + 0.5f) * binw;
      float hub = (r <= DELTA_) ? 0.5f * r * r / DELTA_ : r - 0.5f * DELTA_;
      float take = fminf(c, fmaxf((float)k - cum, 0.f));
      ssum += take * hub;
      cum += c;
      if (cum >= (float)k) break;
    }
    loss = ssum / (float)k;
  }
  ws[OFF_BLOSS + b * 2 + which] = loss;
}

// ---------------- K7: finalize scalars ----------------
__global__ void k7_final(const float* __restrict__ ws, const float* __restrict__ spar,
                         float* __restrict__ out) {
  if (threadIdx.x != 0) return;
  float ld = 0.f, la = 0.f, rs = 0.f;
  for (int b = 0; b < BB; ++b) {
    ld += ws[OFF_BLOSS + b * 2 + 0];
    la += ws[OFF_BLOSS + b * 2 + 1];
    rs += spar[b * 8 + 4];
  }
  ld /= (float)BB; la /= (float)BB; rs /= (float)BB;
  out[0] = ld + LAMB * la;
  out[1] = ld;
  out[2] = la;
  out[3] = rs;
}

extern "C" void kernel_launch(void* const* d_in, const int* in_sizes, int n_in,
                              void* d_out, int out_size, void* d_ws, size_t ws_size,
                              hipStream_t stream) {
  const float* pred = (const float*)d_in[0];
  const float* gt   = (const float*)d_in[1];
  const float* mask = (const float*)d_in[2];
  const float* prev = (const float*)d_in[3];
  float* out = (float*)d_out;
  float* ws = (float*)d_ws;
  unsigned* wsu = (unsigned*)d_ws;
  float* spar = ws + OFF_SPAR;

  k0_init<<<(WS_WORDS + NTHR - 1) / NTHR, NTHR, 0, stream>>>(wsu);
  k1_sums<<<dim3(NBLK, BB), NTHR, 0, stream>>>(pred, gt, mask, ws, wsu);
  k2_solve<<<1, 32, 0, stream>>>(ws, wsu, spar);
  k3_hist_gt<<<dim3(NBLK, BB), NTHR, 0, stream>>>(gt, mask, wsu, spar);
  k4_range<<<BB, 32, 0, stream>>>(ws, wsu, spar);
  k5_resid<<<dim3(NBLK, BB), NTHR, 0, stream>>>(pred, gt, mask, prev, spar, wsu, out + 4);
  k6_huber<<<dim3(2, BB), 32, 0, stream>>>(wsu, ws);
  k7_final<<<1, 32, 0, stream>>>(ws, spar, out);
}
